// Sparsify2D_all_987842478200
// MI455X (gfx1250) — compile-verified, matrix operand order checked by reading, not run on hardware
//
#include <hip/hip_runtime.h>
#include <math.h>

typedef __attribute__((ext_vector_type(4)))  float    v4f;
typedef __attribute__((ext_vector_type(8)))  float    v8f;
typedef __attribute__((ext_vector_type(16))) _Float16 v16h;

// Problem shape is fixed by the reference.
#define BATCH 32
#define CHAN  512
#define HW    3136      // 56*56
#define HW4   784       // HW / 4 (float4 count, exact)

// ---------------------------------------------------------------------------
// Kernel 1: per-(b,c) spatial max over HW elements. One block per channel.
// Reads 205 MB once (vector b128 loads), writes 64 KB.
// ---------------------------------------------------------------------------
__global__ void __launch_bounds__(256)
spatial_max_kernel(const float* __restrict__ x, float* __restrict__ chanmax) {
    __shared__ float red[256];
    const int bc  = blockIdx.x;
    const int tid = threadIdx.x;
    const v4f* p = (const v4f*)(x + (size_t)bc * HW);

    float m = -INFINITY;
    for (int i = tid; i < HW4; i += 256) {
        v4f t = p[i];
        m = fmaxf(m, fmaxf(fmaxf(t.x, t.y), fmaxf(t.z, t.w)));
    }
    red[tid] = m;
    __syncthreads();
    #pragma unroll
    for (int s = 128; s > 0; s >>= 1) {
        if (tid < s) red[tid] = fmaxf(red[tid], red[tid + s]);
        __syncthreads();
    }
    if (tid == 0) chanmax[bc] = red[0];
}

// ---------------------------------------------------------------------------
// Kernel 2: exact top-k selection via WMMA rank counting.
// One block per sample. rank[c] = #{c' beating c (ties broken by index)}.
// Indicator matrix (512x512, binary) x ones-vector done as 16x16x32 f16 WMMA
// tiles with f32 accumulation (counts <= 511, exact). mask[c] = rank[c] < k.
// ---------------------------------------------------------------------------
__global__ void __launch_bounds__(256)
rank_mask_kernel(const float* __restrict__ chanmax, const int* __restrict__ kptr,
                 float* __restrict__ mask) {
    __shared__ float v[CHAN];
    const int b   = blockIdx.x;
    const int tid = threadIdx.x;
    const int k   = *kptr;

    for (int i = tid; i < CHAN; i += 256) v[i] = chanmax[b * CHAN + i];
    __syncthreads();

    const int wave = tid >> 5;
    const int lane = tid & 31;
    const int half = lane >> 4;   // 0: K in {0..7,16..23}; 1: K in {8..15,24..31}
    const int m    = lane & 15;   // A-matrix row within the 16-row tile

    v16h ones;
    #pragma unroll
    for (int j = 0; j < 16; ++j) ones[j] = (_Float16)1.0f;

    // 32 row-tiles of 16 channels; 8 waves -> 4 tiles per wave.
    for (int t = wave; t < CHAN / 16; t += 8) {
        const int base = t * 16;
        const int c    = base + m;
        const float vc = v[c];

        v8f acc = {};
        for (int kk0 = 0; kk0 < CHAN; kk0 += 32) {
            v16h a;
            #pragma unroll
            for (int j = 0; j < 16; ++j) {
                // ISA 16-bit A 16x32 layout: lanes 0-15 hold K 0..7,16..23;
                // lanes 16-31 hold K 8..15,24..31 (VGPR j/2, half j&1).
                const int kof = (j < 8) ? j : (j + 8);
                const int kg  = kk0 + kof + half * 8;
                const float vk = v[kg];
                const bool beat = (vk > vc) || (vk == vc && kg < c);
                a[j] = beat ? (_Float16)1.0f : (_Float16)0.0f;
            }
            acc = __builtin_amdgcn_wmma_f32_16x16x32_f16(
                false, a, false, ones, (short)0, acc, false, false);
        }

        // D layout: VGPR r -> row M=r (lanes 0-15) / M=8+r (lanes 16-31),
        // column N = lane&15. All columns identical; lanes 0 & 16 emit mask.
        if (m == 0) {
            #pragma unroll
            for (int r = 0; r < 8; ++r) {
                const int row = base + half * 8 + r;
                mask[b * CHAN + row] = (acc[r] < (float)k) ? 1.0f : 0.0f;
            }
        }
    }
}

// ---------------------------------------------------------------------------
// Kernel 3: apply mask. One block per (b,c). Masked-off channels write zeros
// without re-reading x (halves the re-read traffic). Non-temporal stores keep
// x resident in the 192MB L2 from pass 1.
// ---------------------------------------------------------------------------
__global__ void __launch_bounds__(256)
apply_mask_kernel(const float* __restrict__ x, const float* __restrict__ mask,
                  float* __restrict__ out) {
    const int bc  = blockIdx.x;
    const int tid = threadIdx.x;
    const float mval = mask[bc];           // block-uniform
    const v4f* src = (const v4f*)(x   + (size_t)bc * HW);
    v4f*       dst = (v4f*)      (out + (size_t)bc * HW);

    if (mval != 0.0f) {
        for (int i = tid; i < HW4; i += 256) {
            v4f t = src[i];
            __builtin_nontemporal_store(t, &dst[i]);
        }
    } else {
        const v4f z = {0.0f, 0.0f, 0.0f, 0.0f};
        for (int i = tid; i < HW4; i += 256) {
            __builtin_nontemporal_store(z, &dst[i]);
        }
    }
}

// ---------------------------------------------------------------------------
extern "C" void kernel_launch(void* const* d_in, const int* in_sizes, int n_in,
                              void* d_out, int out_size, void* d_ws, size_t ws_size,
                              hipStream_t stream) {
    const float* x    = (const float*)d_in[0];
    const int*   kptr = (const int*)d_in[1];
    float*       out  = (float*)d_out;

    float* chanmax = (float*)d_ws;             // [BATCH*CHAN] floats (64 KB)
    float* mask    = chanmax + BATCH * CHAN;   // [BATCH*CHAN] floats (64 KB)

    spatial_max_kernel<<<BATCH * CHAN, 256, 0, stream>>>(x, chanmax);
    rank_mask_kernel  <<<BATCH,        256, 0, stream>>>(chanmax, kptr, mask);
    apply_mask_kernel <<<BATCH * CHAN, 256, 0, stream>>>(x, mask, out);
}
